// DiSA_90812788507199
// MI455X (gfx1250) — matrix operand order repaired and checked by reading, not compiled, not run on hardware
//
#include <hip/hip_runtime.h>
#include <hip/hip_bf16.h>

typedef __attribute__((ext_vector_type(16))) _Float16 v16h;
typedef __attribute__((ext_vector_type(8)))  float    v8f;

#define LOG2E 1.4426950408889634f

// Problem is fixed-shape: B=4, L=256, d_e=d_h=256.
constexpr int cB = 4, cL = 256, cH = 256;
constexpr int cM = cB * cL;     // 1024 GEMM rows
constexpr int cN = cH;          // 256
constexpr int cK = cH;          // 256

// ---------------------------------------------------------------------------
// WMMA GEMM, 16x64 tile per wave (4 N-subtiles reuse one A fragment):
//   out = EPI( A1 @ W1^T (+ A2 @ W2^T) + bias )
// A: (M,K) row-major f32, W: (N,K) row-major f32  =>  result = A @ W^T.
// K fully unrolled: 8 K-steps x 4 subtiles = 32 v_wmma_f32_16x16x32_f16/pass.
// EPI: 0 = none, 1 = ELU, 2 = dual-GEMM sigmoid gate (writes final output).
// ---------------------------------------------------------------------------
template<int EPI>
__device__ __forceinline__
void gemm_pass(const float* __restrict__ A, const float* __restrict__ W,
               int tm, int tn0, int lm, int half, v8f c[4])
{
    const float* pa = A + (size_t)(tm + lm) * cK + half * 16;
    #pragma unroll
    for (int k0 = 0; k0 < cK; k0 += 32) {
        v16h a;
        #pragma unroll
        for (int t = 0; t < 8; ++t) {
            a[t]     = (_Float16)pa[k0 + t];        // K = k0 + half*16 + t
            a[8 + t] = (_Float16)pa[k0 + 8 + t];    // K = k0 + half*16 + 8 + t
        }
        #pragma unroll
        for (int nn = 0; nn < 4; ++nn) {
            const float* pb = W + (size_t)(tn0 + nn * 16 + lm) * cK + half * 16;
            v16h b;
            #pragma unroll
            for (int t = 0; t < 16; ++t)
                b[t] = (_Float16)pb[k0 + t];        // K = k0 + half*16 + t
            c[nn] = __builtin_amdgcn_wmma_f32_16x16x32_f16(false, a, false, b,
                                                           (short)0, c[nn],
                                                           false, false);
        }
    }
}

template<int EPI>
__global__ __launch_bounds__(256)
void gemm_wmma_kernel(const float* __restrict__ A1, const float* __restrict__ W1,
                      const float* __restrict__ A2, const float* __restrict__ W2,
                      const float* __restrict__ bias,
                      const float* __restrict__ rep,  // EPI==2 only
                      const float* __restrict__ attn, // EPI==2 only
                      const float* __restrict__ rmask,// EPI==2 only, (B*L,)
                      float* __restrict__ out)
{
    const int lane = threadIdx.x & 31;
    const int wave = threadIdx.x >> 5;
    const int gw   = blockIdx.x * 8 + wave;       // 256 waves total
    constexpr int stripsN = cN / 64;              // 4
    const int tm   = (gw / stripsN) << 4;         // 16-row tile
    const int tn0  = (gw % stripsN) << 6;         // 64-col strip
    if (tm >= cM) return;

    const int half = lane >> 4;
    const int lm   = lane & 15;

    v8f c[4];
    #pragma unroll
    for (int nn = 0; nn < 4; ++nn)
        c[nn] = v8f{0.f,0.f,0.f,0.f,0.f,0.f,0.f,0.f};

    gemm_pass<EPI>(A1, W1, tm, tn0, lm, half, c);
    if (EPI == 2)
        gemm_pass<EPI>(A2, W2, tm, tn0, lm, half, c);

    // Epilogue. C/D layout: row = tm + 8*half + r, col = tn0 + nn*16 + (lane&15)
    #pragma unroll
    for (int nn = 0; nn < 4; ++nn) {
        const int col = tn0 + nn * 16 + lm;
        const float bv = bias ? bias[col] : 0.0f;
        #pragma unroll
        for (int r = 0; r < 8; ++r) {
            const int row = tm + (half << 3) + r;
            float v = c[nn][r] + bv;
            float res;
            if (EPI == 0) {
                res = v;
            } else if (EPI == 1) {                   // ELU
                res = (v > 0.0f) ? v
                                 : (__builtin_amdgcn_exp2f(v * LOG2E) - 1.0f);
            } else {                                 // sigmoid gate + blend
                const float g  = 1.0f / (1.0f + __builtin_amdgcn_exp2f(-v * LOG2E));
                const float rm = rep [(size_t)row * cN + col];
                const float at = attn[(size_t)row * cN + col];
                res = (g * rm + (1.0f - g) * at) * rmask[row];
            }
            out[(size_t)row * cN + col] = res;
        }
    }
}

// ---------------------------------------------------------------------------
// Fused forward-masked per-channel attention, single streaming pass.
// z = C*tanh((dep[b,j,h] + headb[b,i,h]) / C) is bounded by |C| = 5, so no
// max-subtraction is needed in fp32 and softmax reduces to one pass:
//   attn[b,i,h] = (sum_{j>i} m_j e^{z_j} rep[b,j,h]) / (sum_{j>i} m_j e^{z_j} + 1e-20)
// Block = one (b,i); thread = channel h; rows read coalesced, L2-resident.
// Native gfx1250 trans pipe: v_tanh_f32 + v_exp_f32. Unroll x4 gives the
// scheduler independent streams to cover trans-op latency.
// ---------------------------------------------------------------------------
__global__ __launch_bounds__(256)
void disa_attn_kernel(const float* __restrict__ dep,
                      const float* __restrict__ headb,  // head + b1 pre-folded
                      const float* __restrict__ rep,
                      const float* __restrict__ rmask,  // (B*L,)
                      float* __restrict__ attn)
{
    const int b = blockIdx.x / cL;
    const int i = blockIdx.x % cL;
    const int h = threadIdx.x;
    const size_t base = (size_t)b * cL * cH;

    const float hb     = headb[base + (size_t)i * cH + h];
    const float invC   = 1.0f / 5.0f;
    const float Clog2e = 5.0f * LOG2E;

    float sum = 0.0f, acc = 0.0f;
    #pragma unroll 4
    for (int j = i + 1; j < cL; ++j) {
        const float mk = rmask[b * cL + j];
        const float d  = dep[base + (size_t)j * cH + h];
        const float r  = rep[base + (size_t)j * cH + h];
        const float a  = (d + hb) * invC;
        float t;
#if __has_builtin(__builtin_amdgcn_tanhf)
        t = __builtin_amdgcn_tanhf(a);
#else
        // trailing v_nop satisfies the 1-independent-op trans hazard
        asm volatile("v_tanh_f32 %0, %1\n\tv_nop" : "=v"(t) : "v"(a));
#endif
        const float e = mk * __builtin_amdgcn_exp2f(t * Clog2e);
        sum += e;
        acc += e * r;
    }
    // Reference: sums==0 -> +1 -> 0/1 = 0; here acc==0 too, so identical.
    attn[base + (size_t)i * cH + h] = acc / (sum + 1e-20f);
}

// ---------------------------------------------------------------------------
extern "C" void kernel_launch(void* const* d_in, const int* in_sizes, int n_in,
                              void* d_out, int out_size, void* d_ws, size_t ws_size,
                              hipStream_t stream)
{
    const float* x     = (const float*)d_in[0];  // (B,L,256)
    const float* rmask = (const float*)d_in[1];  // (B,L,1)
    const float* fc_w  = (const float*)d_in[2];  // (256,256)
    const float* fc_b  = (const float*)d_in[3];  // (256,)
    const float* w1    = (const float*)d_in[4];
    const float* w2    = (const float*)d_in[5];
    const float* b1    = (const float*)d_in[6];
    const float* wf1   = (const float*)d_in[7];
    const float* wf2   = (const float*)d_in[8];
    const float* bf    = (const float*)d_in[9];
    float* out = (float*)d_out;

    float* ws     = (float*)d_ws;
    float* repmap = ws;                 // cM*cN floats (1 MB)
    float* dep    = ws + 1 * (cM * cN);
    float* headb  = ws + 2 * (cM * cN);
    float* attn   = ws + 3 * (cM * cN);

    const dim3 gblk(256);
    const dim3 ggrd(((cM / 16) * (cN / 64)) / 8);   // 256 wave-tiles / 8 waves

    // 1) rep_map = elu(x @ fc_w^T + fc_b)
    gemm_wmma_kernel<1><<<ggrd, gblk, 0, stream>>>(
        x, fc_w, nullptr, nullptr, fc_b, nullptr, nullptr, nullptr, repmap);

    // 2) dep = rep_map @ w1^T
    gemm_wmma_kernel<0><<<ggrd, gblk, 0, stream>>>(
        repmap, w1, nullptr, nullptr, nullptr, nullptr, nullptr, nullptr, dep);

    // 3) headb = rep_map @ w2^T + b1   (b1 folded)
    gemm_wmma_kernel<0><<<ggrd, gblk, 0, stream>>>(
        repmap, w2, nullptr, nullptr, b1, nullptr, nullptr, nullptr, headb);

    // 4) fused masked per-channel softmax-attention (never materializes B*L*L*H)
    disa_attn_kernel<<<dim3(cB * cL), dim3(cH), 0, stream>>>(
        dep, headb, repmap, rmask, attn);

    // 5) gate = sigmoid(rep@wf1^T + attn@wf2^T + bf); out = blend * rep_mask
    gemm_wmma_kernel<2><<<ggrd, gblk, 0, stream>>>(
        repmap, wf1, attn, wf2, bf, repmap, attn, rmask, out);
}
